// Attention_89593017794629
// MI455X (gfx1250) — matrix operand order repaired and checked by reading, not compiled, not run on hardware
//
#include <hip/hip_runtime.h>

// ---------------------------------------------------------------------------
// Types for CDNA5 WMMA (wave32): v_wmma_f32_16x16x32_bf16
// ---------------------------------------------------------------------------
typedef __bf16 bf16;
typedef __attribute__((ext_vector_type(16))) __bf16 v16bf;
typedef __attribute__((ext_vector_type(8)))  __bf16 v8bf;
typedef __attribute__((ext_vector_type(8)))  float  v8f;

#define B_  2
#define S_  2048
#define D_  1024
#define H_  16
#define DH_ 64
#define INNER_ 1024
#define N3_ 3072
#define SCALE_ 0.125f   // 64^-0.5
#define LN_EPS_ 1e-5f

__device__ __forceinline__ v8f wmma_bf16(v16bf a, v16bf b, v8f c) {
  return __builtin_amdgcn_wmma_f32_16x16x32_bf16(
      /*neg_a=*/false, a, /*neg_b=*/false, b,
      /*c_mod=*/(short)0, c, /*reuse_a=*/false, /*reuse_b=*/false);
}

// A-matrix 16x32 bf16 fragment (ISA 7.12.2): lane row = lane%16,
// elements 0..7 = K[klow..klow+7], 8..15 = K[klow+16..klow+23], klow=8*(lane/16)
__device__ __forceinline__ v16bf frag_a(const bf16* prow, int klow) {
  v8bf lo = *(const v8bf*)(prow + klow);
  v8bf hi = *(const v8bf*)(prow + klow + 16);
  v16bf r;
#pragma unroll
  for (int i = 0; i < 8; ++i) { r[i] = lo[i]; r[i + 8] = hi[i]; }
  return r;
}

// B-matrix 32x16 bf16 fragment: lane col = lane%16,
// elements 0..7 = K[kb16/2..kb16/2+7], 8..15 = K[kb16/2+16..kb16/2+23], kb16 = 16*(lane/16)
__device__ __forceinline__ v16bf frag_b(const bf16* prow, int kb16) {
  v8bf lo = *(const v8bf*)(prow + (kb16 >> 1));
  v8bf hi = *(const v8bf*)(prow + (kb16 >> 1) + 16);
  v16bf r;
#pragma unroll
  for (int i = 0; i < 8; ++i) { r[i] = lo[i]; r[i + 8] = hi[i]; }
  return r;
}

// 16-byte global -> LDS copy on the CDNA5 async path (ASYNCcnt tracked),
// via inline asm (ISA 15.18.3 opcode 98: GLOBAL_LOAD_ASYNC_TO_LDS_B128,
// GV mode: vdst = LDS byte address VGPR, vaddr = 64-bit global address).
__device__ __forceinline__ void cp16_async(const bf16* g, bf16* l) {
#if defined(__gfx1250__)
  const unsigned lds_addr =
      (unsigned)(unsigned long long)(__attribute__((address_space(3))) void*)(void*)l;
  const unsigned long long gaddr = (unsigned long long)g;
  asm volatile("global_load_async_to_lds_b128 %0, %1, off"
               :: "v"(lds_addr), "v"(gaddr) : "memory");
#else
  *(v8bf*)l = *(const v8bf*)g;
#endif
}

__device__ __forceinline__ void async_wait0() {
#if defined(__gfx1250__)
#if __has_builtin(__builtin_amdgcn_s_wait_asynccnt)
  __builtin_amdgcn_s_wait_asynccnt(0);
#else
  asm volatile("s_wait_asynccnt 0x0" ::: "memory");
#endif
#endif
}

// ---------------------------------------------------------------------------
// Kernel 1: LayerNorm (f32 in) -> bf16 out.  One 256-thread block per row.
// ---------------------------------------------------------------------------
__global__ void k_layernorm(const float* __restrict__ x,
                            const float* __restrict__ gamma,
                            const float* __restrict__ beta,
                            bf16* __restrict__ xn) {
  const int row  = blockIdx.x;
  const int lane = threadIdx.x & 31;
  const int wid  = threadIdx.x >> 5;
  const float* xr = x + (size_t)row * D_;

  float s = 0.f, s2 = 0.f;
  for (int i = threadIdx.x; i < D_; i += 256) {
    float v = xr[i];
    s += v; s2 += v * v;
  }
#pragma unroll
  for (int off = 16; off; off >>= 1) {
    s  += __shfl_xor(s, off);
    s2 += __shfl_xor(s2, off);
  }
  __shared__ float sh[16];
  if (lane == 0) { sh[wid] = s; sh[8 + wid] = s2; }
  __syncthreads();
  float ts = 0.f, ts2 = 0.f;
#pragma unroll
  for (int i = 0; i < 8; ++i) { ts += sh[i]; ts2 += sh[8 + i]; }
  const float mean = ts * (1.0f / D_);
  const float var  = ts2 * (1.0f / D_) - mean * mean;
  const float rstd = rsqrtf(var + LN_EPS_);

  bf16* xo = xn + (size_t)row * D_;
  for (int i = threadIdx.x; i < D_; i += 256) {
    float v = (xr[i] - mean) * rstd * gamma[i] + beta[i];
    xo[i] = (bf16)v;
  }
}

// ---------------------------------------------------------------------------
// Kernel 2: transpose + convert  f32 [R][C] -> bf16 [C][R]  (tiled via LDS)
// ---------------------------------------------------------------------------
__global__ void k_transpose_bf16(const float* __restrict__ in,
                                 bf16* __restrict__ out, int R, int C) {
  __shared__ float t[32][33];
  const int c0 = blockIdx.x * 32;
  const int r0 = blockIdx.y * 32;
  for (int i = threadIdx.y; i < 32; i += 8)
    t[i][threadIdx.x] = in[(size_t)(r0 + i) * C + (c0 + threadIdx.x)];
  __syncthreads();
  for (int i = threadIdx.y; i < 32; i += 8)
    out[(size_t)(c0 + i) * R + (r0 + threadIdx.x)] = (bf16)t[threadIdx.x][i];
}

// ---------------------------------------------------------------------------
// Kernel 3: QKV GEMM  xn[4096,1024] @ wqkvT[3072,1024]^T  (both bf16)
// 8 waves/block, each wave owns a 32x64 tile (2x4 WMMA tiles, 8 WMMA per
// 6 fragment loads); block tile = 32(M) x 512(N).
// Epilogue scatters into Q [b,h,s,dh], K [b,h,s,dh], Vt [b,h,dh,s] (bf16).
// ---------------------------------------------------------------------------
__device__ __forceinline__ void store_qkv_tile(const v8f& c, int mbase, int nbase,
                                               int lane, bf16* q, bf16* k, bf16* v) {
  const int half = lane >> 4;
  const int n = nbase + (lane & 15);
  const int sec = n >> 10;          // 0:Q 1:K 2:V
  const int cc = n & 1023;
  const int h = cc >> 6, dc = cc & 63;
#pragma unroll
  for (int r = 0; r < 8; ++r) {
    const int m = mbase + r + 8 * half;       // row in [0, 4096)
    const int bb = m >> 11, ss = m & 2047;
    const bf16 val = (bf16)c[r];
    const size_t hb = (size_t)bb * H_ + h;
    if (sec == 0)      q[(hb * S_ + ss) * DH_ + dc] = val;
    else if (sec == 1) k[(hb * S_ + ss) * DH_ + dc] = val;
    else               v[(hb * DH_ + dc) * S_ + ss] = val;
  }
}

__global__ void k_gemm_qkv(const bf16* __restrict__ xn,
                           const bf16* __restrict__ wT,   // [3072][1024]
                           bf16* __restrict__ q, bf16* __restrict__ k,
                           bf16* __restrict__ v) {
  const int lane = threadIdx.x & 31;
  const int wid  = threadIdx.x >> 5;
  const int m0 = blockIdx.y * 32;
  const int n0 = blockIdx.x * 512 + wid * 64;
  const int klow = 8 * (lane >> 4);
  const int kb16 = 16 * (lane >> 4);
  const int ln16 = lane & 15;

  const bf16* arow0 = xn + (size_t)(m0 + ln16) * D_;
  const bf16* arow1 = arow0 + (size_t)16 * D_;
  const bf16* brow0 = wT + (size_t)(n0 + ln16) * D_;
  const bf16* brow1 = brow0 + (size_t)16 * D_;
  const bf16* brow2 = brow0 + (size_t)32 * D_;
  const bf16* brow3 = brow0 + (size_t)48 * D_;

  v8f c[2][4] = {};
  for (int kk = 0; kk < D_; kk += 32) {
    v16bf a0 = frag_a(arow0 + kk, klow);
    v16bf a1 = frag_a(arow1 + kk, klow);
    v16bf b0 = frag_b(brow0 + kk, kb16);
    v16bf b1 = frag_b(brow1 + kk, kb16);
    v16bf b2 = frag_b(brow2 + kk, kb16);
    v16bf b3 = frag_b(brow3 + kk, kb16);
    c[0][0] = wmma_bf16(a0, b0, c[0][0]);
    c[0][1] = wmma_bf16(a0, b1, c[0][1]);
    c[0][2] = wmma_bf16(a0, b2, c[0][2]);
    c[0][3] = wmma_bf16(a0, b3, c[0][3]);
    c[1][0] = wmma_bf16(a1, b0, c[1][0]);
    c[1][1] = wmma_bf16(a1, b1, c[1][1]);
    c[1][2] = wmma_bf16(a1, b2, c[1][2]);
    c[1][3] = wmma_bf16(a1, b3, c[1][3]);
  }
#pragma unroll
  for (int i = 0; i < 2; ++i)
#pragma unroll
    for (int j = 0; j < 4; ++j)
      store_qkv_tile(c[i][j], m0 + 16 * i, n0 + 16 * j, lane, q, k, v);
}

// ---------------------------------------------------------------------------
// Kernel 4: flash attention.  grid = (b*h, s/128), block = 256 (8 waves).
// Each wave owns 16 query rows; loops over keys in chunks of 32.
// K and Vt chunks are staged once per block into LDS through the async
// global->LDS path (ASYNCcnt) and served to all 8 waves from there.
// ---------------------------------------------------------------------------
__global__ void k_flash_attn(const bf16* __restrict__ q,
                             const bf16* __restrict__ k,
                             const bf16* __restrict__ v,   // [b,h,dh,s]
                             bf16* __restrict__ attn) {    // [b,s,inner]
  const int lane = threadIdx.x & 31;
  const int wid  = threadIdx.x >> 5;
  const int bh   = blockIdx.x;                 // b*16 + h
  const int q0   = blockIdx.y * 128 + wid * 16;
  const int klow = 8 * (lane >> 4);
  const int kb16 = 16 * (lane >> 4);
  const int ln16 = lane & 15;
  const int half = lane >> 4;

  const size_t base = (size_t)bh * S_ * DH_;
  const bf16* Q = q + base;
  const bf16* K = k + base;
  const bf16* V = v + base;                    // [64][2048]

  const bf16* qrow = Q + (size_t)(q0 + ln16) * DH_;
  const v16bf a0 = frag_a(qrow,      klow);    // dh 0..31
  const v16bf a1 = frag_a(qrow + 32, klow);    // dh 32..63

  float Mrow[8], Lrow[8], al[8];
  v8f o0 = {}, o1 = {}, o2 = {}, o3 = {};
#pragma unroll
  for (int r = 0; r < 8; ++r) { Mrow[r] = -__builtin_inff(); Lrow[r] = 0.f; }

  __shared__ bf16 Kl[32][64];                  // keys kc..kc+31 (row-major)
  __shared__ bf16 Vl[64][32];                  // Vt rows (dh) x 32 keys
  __shared__ bf16 pl[8][16][32];               // per-wave P staging

  const int t = threadIdx.x;
  const int vr = t >> 2, vs = (t & 3) * 8;

  for (int kc = 0; kc < S_; kc += 32) {
    // --- cooperative async stage of K (4KB contiguous) and Vt (64x64B) ---
    cp16_async(K + (size_t)kc * DH_ + t * 8, &Kl[0][0] + t * 8);
    cp16_async(V + (size_t)vr * S_ + kc + vs, &Vl[vr][vs]);
    async_wait0();
    __syncthreads();

    // --- S = Q K^T (served from LDS) ---
    const bf16* kr0 = &Kl[ln16][0];
    const bf16* kr1 = &Kl[16 + ln16][0];
    v8f s0 = {}, s1 = {};
    s0 = wmma_bf16(a0, frag_b(kr0,      kb16), s0);
    s0 = wmma_bf16(a1, frag_b(kr0 + 32, kb16), s0);
    s1 = wmma_bf16(a0, frag_b(kr1,      kb16), s1);
    s1 = wmma_bf16(a1, frag_b(kr1 + 32, kb16), s1);

#pragma unroll
    for (int r = 0; r < 8; ++r) { s0[r] *= SCALE_; s1[r] *= SCALE_; }

    // --- online softmax (row reductions across the 16-lane half) ---
#pragma unroll
    for (int r = 0; r < 8; ++r) {
      float mx = fmaxf(s0[r], s1[r]);
#pragma unroll
      for (int off = 8; off; off >>= 1) mx = fmaxf(mx, __shfl_xor(mx, off));
      const float mn = fmaxf(Mrow[r], mx);
      const float alpha = __expf(Mrow[r] - mn);
      const float p0 = __expf(s0[r] - mn);
      const float p1 = __expf(s1[r] - mn);
      s0[r] = p0; s1[r] = p1;
      float rs = p0 + p1;
#pragma unroll
      for (int off = 8; off; off >>= 1) rs += __shfl_xor(rs, off);
      Lrow[r] = Lrow[r] * alpha + rs;
      Mrow[r] = mn;
      al[r] = alpha;
    }
#pragma unroll
    for (int r = 0; r < 8; ++r) {
      o0[r] *= al[r]; o1[r] *= al[r]; o2[r] *= al[r]; o3[r] *= al[r];
    }

    // --- C-layout -> A-layout shuffle of P through per-wave LDS ---
#pragma unroll
    for (int r = 0; r < 8; ++r) {
      pl[wid][r + 8 * half][ln16]      = (bf16)s0[r];
      pl[wid][r + 8 * half][16 + ln16] = (bf16)s1[r];
    }
    asm volatile("s_wait_dscnt 0x0" ::: "memory");
    const v16bf pf = frag_a(&pl[wid][ln16][0], klow);

    // --- O += P V  (V fragments from LDS) ---
    o0 = wmma_bf16(pf, frag_b(&Vl[     ln16][0], kb16), o0);
    o1 = wmma_bf16(pf, frag_b(&Vl[16 + ln16][0], kb16), o1);
    o2 = wmma_bf16(pf, frag_b(&Vl[32 + ln16][0], kb16), o2);
    o3 = wmma_bf16(pf, frag_b(&Vl[48 + ln16][0], kb16), o3);

    __syncthreads();   // protect Kl/Vl before next chunk overwrites them
  }

  const int bb = bh >> 4, h = bh & 15;
#pragma unroll
  for (int r = 0; r < 8; ++r) {
    const float inv = 1.f / Lrow[r];
    const int m = q0 + r + 8 * half;
    const size_t rowoff = ((size_t)bb * S_ + m) * INNER_ + h * DH_;
    attn[rowoff +      ln16] = (bf16)(o0[r] * inv);
    attn[rowoff + 16 + ln16] = (bf16)(o1[r] * inv);
    attn[rowoff + 32 + ln16] = (bf16)(o2[r] * inv);
    attn[rowoff + 48 + ln16] = (bf16)(o3[r] * inv);
  }
}

// ---------------------------------------------------------------------------
// Kernel 5: output GEMM  attn[4096,1024] @ woutT[1024,1024]^T + bias -> f32
// Same 32x64 wave tile as the QKV GEMM.
// ---------------------------------------------------------------------------
__device__ __forceinline__ void store_out_tile(const v8f& c, int mbase, int nbase,
                                               int lane, const float* bias,
                                               float* out) {
  const int half = lane >> 4;
  const int n = nbase + (lane & 15);
  const float bv = bias[n];
#pragma unroll
  for (int r = 0; r < 8; ++r) {
    const int m = mbase + r + 8 * half;
    out[(size_t)m * D_ + n] = c[r] + bv;
  }
}

__global__ void k_gemm_out(const bf16* __restrict__ attn,
                           const bf16* __restrict__ wT,    // [1024][1024]
                           const float* __restrict__ bias,
                           float* __restrict__ out) {
  const int lane = threadIdx.x & 31;
  const int wid  = threadIdx.x >> 5;
  const int m0 = blockIdx.y * 32;
  const int n0 = blockIdx.x * 512 + wid * 64;
  const int klow = 8 * (lane >> 4);
  const int kb16 = 16 * (lane >> 4);
  const int ln16 = lane & 15;

  const bf16* arow0 = attn + (size_t)(m0 + ln16) * INNER_;
  const bf16* arow1 = arow0 + (size_t)16 * INNER_;
  const bf16* brow0 = wT + (size_t)(n0 + ln16) * INNER_;
  const bf16* brow1 = brow0 + (size_t)16 * INNER_;
  const bf16* brow2 = brow0 + (size_t)32 * INNER_;
  const bf16* brow3 = brow0 + (size_t)48 * INNER_;

  v8f c[2][4] = {};
  for (int kk = 0; kk < INNER_; kk += 32) {
    v16bf a0 = frag_a(arow0 + kk, klow);
    v16bf a1 = frag_a(arow1 + kk, klow);
    v16bf b0 = frag_b(brow0 + kk, kb16);
    v16bf b1 = frag_b(brow1 + kk, kb16);
    v16bf b2 = frag_b(brow2 + kk, kb16);
    v16bf b3 = frag_b(brow3 + kk, kb16);
    c[0][0] = wmma_bf16(a0, b0, c[0][0]);
    c[0][1] = wmma_bf16(a0, b1, c[0][1]);
    c[0][2] = wmma_bf16(a0, b2, c[0][2]);
    c[0][3] = wmma_bf16(a0, b3, c[0][3]);
    c[1][0] = wmma_bf16(a1, b0, c[1][0]);
    c[1][1] = wmma_bf16(a1, b1, c[1][1]);
    c[1][2] = wmma_bf16(a1, b2, c[1][2]);
    c[1][3] = wmma_bf16(a1, b3, c[1][3]);
  }
#pragma unroll
  for (int i = 0; i < 2; ++i)
#pragma unroll
    for (int j = 0; j < 4; ++j)
      store_out_tile(c[i][j], m0 + 16 * i, n0 + 16 * j, lane, bias, out);
}

// ---------------------------------------------------------------------------
// Host launcher
// ---------------------------------------------------------------------------
extern "C" void kernel_launch(void* const* d_in, const int* in_sizes, int n_in,
                              void* d_out, int out_size, void* d_ws, size_t ws_size,
                              hipStream_t stream) {
  const float* x     = (const float*)d_in[0];
  const float* gamma = (const float*)d_in[1];
  const float* beta  = (const float*)d_in[2];
  const float* wqkv  = (const float*)d_in[3];
  const float* wout  = (const float*)d_in[4];
  const float* bout  = (const float*)d_in[5];
  float* out = (float*)d_out;

  char* ws = (char*)d_ws;
  const size_t ROWS = (size_t)B_ * S_;                 // 4096
  bf16* xn   = (bf16*)ws;  ws += ROWS * D_ * sizeof(bf16);          // 8 MB
  bf16* wqT  = (bf16*)ws;  ws += (size_t)N3_ * D_ * sizeof(bf16);   // 6 MB
  bf16* woT  = (bf16*)ws;  ws += (size_t)D_ * INNER_ * sizeof(bf16);// 2 MB
  bf16* qb   = (bf16*)ws;  ws += ROWS * INNER_ * sizeof(bf16);      // 8 MB
  bf16* kb   = (bf16*)ws;  ws += ROWS * INNER_ * sizeof(bf16);      // 8 MB
  bf16* vtb  = (bf16*)ws;  ws += ROWS * INNER_ * sizeof(bf16);      // 8 MB
  bf16* attn = (bf16*)ws;  ws += ROWS * INNER_ * sizeof(bf16);      // 8 MB

  k_layernorm<<<(int)ROWS, 256, 0, stream>>>(x, gamma, beta, xn);

  k_transpose_bf16<<<dim3(N3_ / 32, D_ / 32), dim3(32, 8), 0, stream>>>(
      wqkv, wqT, D_, N3_);
  k_transpose_bf16<<<dim3(D_ / 32, INNER_ / 32), dim3(32, 8), 0, stream>>>(
      wout, woT, INNER_, D_);

  k_gemm_qkv<<<dim3(N3_ / 512, (int)ROWS / 32), 256, 0, stream>>>(
      xn, wqT, qb, kb, vtb);

  k_flash_attn<<<dim3(B_ * H_, S_ / 128), 256, 0, stream>>>(qb, kb, vtb, attn);

  k_gemm_out<<<dim3(INNER_ / 512, (int)ROWS / 32), 256, 0, stream>>>(
      attn, woT, bout, out);
}
